// PQLayer_69535520522886
// MI455X (gfx1250) — matrix operand order repaired and checked by reading, not compiled, not run on hardware
//
#include <hip/hip_runtime.h>

// Soft Product-Quantization layer, fused, bf16 WMMA on gfx1250 (MI455X).
//   x: (N=65536, D=1024) f32, codebook: (M=8, K=256, D_M=128) f32, temperature: int scalar
//   out: (N, 1024) f32
// Softmax trick: x^2 term cancels; logits = (2*inter - c2)/T.

#define M_SUB 8
#define K_CODES 256
#define D_M 128
#define D_TOT 1024
#define WAVES 8
#define ROWS_PER_BLOCK (16 * WAVES)
#define TILES 4                       // row-tiles per block (amortize codebook->LDS)

// Padded LDS strides (in ushorts) to avoid 64-bank conflicts on strided access.
#define CBN_STRIDE 136   // [code][d]  bf16 (GEMM1 B source)
#define CBT_STRIDE 264   // [d][code]  bf16 (GEMM2 B source)
#define P_CSTRIDE  24    // per-wave P col-major [code][row16] bf16 (GEMM2 A source)

typedef __attribute__((ext_vector_type(16))) __bf16          v16bf;
typedef __attribute__((ext_vector_type(8)))  float           v8f;
typedef __attribute__((ext_vector_type(16))) unsigned short  v16u;
typedef __attribute__((ext_vector_type(8)))  unsigned int    v8ui;
typedef __attribute__((ext_vector_type(4)))  unsigned int    v4ui;
typedef __attribute__((ext_vector_type(2)))  unsigned int    v2ui;

// Pack two f32 -> two bf16 in one u32 using CDNA5 VOP3P mixed-precision FMA
// converts (ISA 15.10 opcodes 62/63): 2 VALU ops per pair, result pre-packed.
__device__ __forceinline__ unsigned int pkbf(float a, float b) {
  unsigned int u;
  asm("v_fma_mixlo_bf16 %0, %1, 1.0, 0\n\t"
      "v_fma_mixhi_bf16 %0, %2, 1.0, 0"
      : "=&v"(u)
      : "v"(a), "v"(b));
  return u;
}

__device__ __forceinline__ float fast_exp2(float a) {
#if __has_builtin(__builtin_amdgcn_exp2f)
  return __builtin_amdgcn_exp2f(a);     // raw v_exp_f32 (input <= 0 here)
#else
  return exp2f(a);
#endif
}

__device__ __forceinline__ float fast_rcp(float a) {
#if __has_builtin(__builtin_amdgcn_rcpf)
  return __builtin_amdgcn_rcpf(a);      // raw v_rcp_f32
#else
  return 1.0f / a;
#endif
}

__global__ void __launch_bounds__(WAVES * 32, 1)
softpq_wmma_kernel(const float* __restrict__ x, const float* __restrict__ cb,
                   const int* __restrict__ temp, float* __restrict__ out)
{
  extern __shared__ char smem_raw[];
  unsigned short* cbN  = (unsigned short*)smem_raw;            // 256*136 ush
  unsigned short* cbT  = cbN + K_CODES * CBN_STRIDE;           // 128*264 ush
  float*          c2s  = (float*)(cbT + D_M * CBT_STRIDE);     // 256 f32
  unsigned short* Pall = (unsigned short*)(c2s + K_CODES);     // 8*256*24 ush

  const int m    = blockIdx.y;
  const int tid  = threadIdx.x;
  const int wave = tid >> 5;
  const int lane = tid & 31;
  const int r    = lane & 15;   // column-in-tile / row-in-A index
  const int hlf  = lane >> 4;   // lane half

  // ---------------- Phase A: codebook_m -> LDS bf16 (two layouts) + c2 ----------------
  {
    const int c = tid;                                  // one codeword per thread
    const float* crow = cb + ((size_t)m * K_CODES + c) * D_M;
    float ss = 0.f;
    #pragma unroll 4
    for (int d = 0; d < D_M; d += 4) {
      float4 v = *(const float4*)(crow + d);
      ss += v.x * v.x + v.y * v.y + v.z * v.z + v.w * v.w;
      unsigned int p0 = pkbf(v.x, v.y);
      unsigned int p1 = pkbf(v.z, v.w);
      v2ui pp = {p0, p1};
      *(v2ui*)(cbN + c * CBN_STRIDE + d) = pp;          // 8B store
      cbT[(d + 0) * CBT_STRIDE + c] = (unsigned short)(p0);
      cbT[(d + 1) * CBT_STRIDE + c] = (unsigned short)(p0 >> 16);
      cbT[(d + 2) * CBT_STRIDE + c] = (unsigned short)(p1);
      cbT[(d + 3) * CBT_STRIDE + c] = (unsigned short)(p1 >> 16);
    }
    c2s[c] = ss;
  }
  __syncthreads();

  const float invT   = 1.0f / (float)temp[0];
  const float scale  = invT * 1.44269504088896341f;   // log2(e)/T -> exp2 domain
  const float scale2 = 2.0f * scale;

  unsigned short* P = Pall + wave * K_CODES * P_CSTRIDE;

  for (int it = 0; it < TILES; ++it) {
    const int n0 = (blockIdx.x * TILES + it) * ROWS_PER_BLOCK + wave * 16;
    const float* xrow = x + (size_t)(n0 + r) * D_TOT + m * D_M;

    // ------------ GEMM1 A fragments: x rows -> bf16, contiguous-K A-layout (B matches) ------------
    // lanes 0-15: K = d0+{0..15}; lanes 16-31: +16 (d0 = 32*s)
    v16bf afrag[4];
    #pragma unroll
    for (int s = 0; s < 4; ++s) {
      const int d0 = 32 * s + 16 * hlf;
      float4 a0 = *(const float4*)(xrow + d0 + 0);
      float4 a1 = *(const float4*)(xrow + d0 + 4);
      float4 b0 = *(const float4*)(xrow + d0 + 8);
      float4 b1 = *(const float4*)(xrow + d0 + 12);
      v8ui t;
      t[0] = pkbf(a0.x, a0.y); t[1] = pkbf(a0.z, a0.w);
      t[2] = pkbf(a1.x, a1.y); t[3] = pkbf(a1.z, a1.w);
      t[4] = pkbf(b0.x, b0.y); t[5] = pkbf(b0.z, b0.w);
      t[6] = pkbf(b1.x, b1.y); t[7] = pkbf(b1.z, b1.w);
      afrag[s] = __builtin_bit_cast(v16bf, t);
    }

    // Prefetch next tile's x rows into cache (global_prefetch_b8).
    if (it + 1 < TILES) {
      const float* xnext = xrow + (size_t)ROWS_PER_BLOCK * D_TOT;
      __builtin_prefetch(xnext + 0, 0, 3);
      __builtin_prefetch(xnext + 32, 0, 3);
      __builtin_prefetch(xnext + 64, 0, 3);
      __builtin_prefetch(xnext + 96, 0, 3);
    }

    // ------------ GEMM1: inter(16 x 256) = X(16x128) * C^T(128x256) ------------
    v8f acc[16];
    #pragma unroll
    for (int t = 0; t < 16; ++t) {
      v8f z = {0.f, 0.f, 0.f, 0.f, 0.f, 0.f, 0.f, 0.f};
      acc[t] = z;
    }
    #pragma unroll
    for (int t = 0; t < 16; ++t) {                 // codeword tile (16 codes)
      const unsigned short* bp = cbN + (t * 16 + r) * CBN_STRIDE;
      #pragma unroll
      for (int s = 0; s < 4; ++s) {                // K-step over d
        // B-layout: lanes 0-15 hold K=d0..d0+15, lanes 16-31 hold K=d0+16..d0+31
        v16u bu = *(const v16u*)(bp + 32 * s + 16 * hlf);
        v16bf bf = __builtin_bit_cast(v16bf, bu);
        acc[t] = __builtin_amdgcn_wmma_f32_16x16x32_bf16(
            false, afrag[s], false, bf, (short)0, acc[t], false, false);
      }
    }

    // ------------ logits + row softmax (in C/D layout) ------------
    // acc[t][v]: row = v + 8*hlf, col = t*16 + r. A row lives in 16 lanes of one half.
    float mx[8];
    #pragma unroll
    for (int v = 0; v < 8; ++v) mx[v] = -1e30f;
    #pragma unroll
    for (int t = 0; t < 16; ++t) {
      const float ncc = -c2s[t * 16 + r] * scale;        // hoisted per tile
      #pragma unroll
      for (int v = 0; v < 8; ++v) {
        float lg = fmaf(acc[t][v], scale2, ncc);         // (2*inter - c2)*scale, 1 FMA
        acc[t][v] = lg;
        mx[v] = fmaxf(mx[v], lg);
      }
    }
    #pragma unroll
    for (int v = 0; v < 8; ++v) {
      #pragma unroll
      for (int mask = 1; mask < 16; mask <<= 1)
        mx[v] = fmaxf(mx[v], __shfl_xor(mx[v], mask, 32));
    }
    float sum[8];
    #pragma unroll
    for (int v = 0; v < 8; ++v) sum[v] = 0.f;
    #pragma unroll
    for (int t = 0; t < 16; ++t) {
      #pragma unroll
      for (int v = 0; v < 8; ++v) {
        float e = fast_exp2(acc[t][v] - mx[v]);
        acc[t][v] = e;
        sum[v] += e;
      }
    }
    #pragma unroll
    for (int v = 0; v < 8; ++v) {
      #pragma unroll
      for (int mask = 1; mask < 16; mask <<= 1)
        sum[v] += __shfl_xor(sum[v], mask, 32);
      sum[v] = fast_rcp(sum[v]);
    }

    // ------------ P -> LDS bf16, col-major [code][row16] (wave-private) ------------
    // One 16B ds_store per tile; banks 12*r mod 64 all distinct -> conflict-free.
    #pragma unroll
    for (int t = 0; t < 16; ++t) {
      v4ui pk;
      #pragma unroll
      for (int v = 0; v < 4; ++v)
        pk[v] = pkbf(acc[t][2 * v] * sum[2 * v], acc[t][2 * v + 1] * sum[2 * v + 1]);
      *(v4ui*)(P + (t * 16 + r) * P_CSTRIDE + 8 * hlf) = pk;
    }
    __builtin_amdgcn_wave_barrier();   // keep LDS stores ordered before reloads (same-wave)

    // ------------ GEMM2: out(16 x 128) = P(16x256) * C(256x128) ------------
    v8f acc2[8];
    #pragma unroll
    for (int t = 0; t < 8; ++t) {
      v8f z = {0.f, 0.f, 0.f, 0.f, 0.f, 0.f, 0.f, 0.f};
      acc2[t] = z;
    }
    const unsigned short* pb = P + r;   // A-gather base: addr = code*P_CSTRIDE + row
    #pragma unroll
    for (int s = 0; s < 8; ++s) {                  // K-step over codes
      // A fragment: row = r, K = c0+{0..15} (lanes 0-15); +16 for high half
      const int c0 = 32 * s + 16 * hlf;
      v16u au;
      #pragma unroll
      for (int j = 0; j < 8; ++j) {
        au[j]     = pb[(c0 + j) * P_CSTRIDE];
        au[8 + j] = pb[(c0 + 8 + j) * P_CSTRIDE];
      }
      v16bf af = __builtin_bit_cast(v16bf, au);
      const int cb0 = 32 * s + 16 * hlf;           // B K-range for this lane half
      #pragma unroll
      for (int t = 0; t < 8; ++t) {                // d tile
        v16u bu = *(const v16u*)(cbT + (t * 16 + r) * CBT_STRIDE + cb0);
        v16bf bf = __builtin_bit_cast(v16bf, bu);
        acc2[t] = __builtin_amdgcn_wmma_f32_16x16x32_bf16(
            false, af, false, bf, (short)0, acc2[t], false, false);
      }
    }

    // ------------ store out (f32). Lanes 0-15 / 16-31 each form a 64B run ------------
    float* orow = out + (size_t)n0 * D_TOT + m * D_M;
    #pragma unroll
    for (int t = 0; t < 8; ++t) {
      const int d = t * 16 + r;
      #pragma unroll
      for (int v = 0; v < 8; ++v) {
        const int row = v + 8 * hlf;
        orow[(size_t)row * D_TOT + d] = acc2[t][v];
      }
    }
  }
}

extern "C" void kernel_launch(void* const* d_in, const int* in_sizes, int n_in,
                              void* d_out, int out_size, void* d_ws, size_t ws_size,
                              hipStream_t stream) {
  const float* x    = (const float*)d_in[0];
  const float* cb   = (const float*)d_in[1];
  const int*   temp = (const int*)d_in[2];
  float*       out  = (float*)d_out;

  const int n = in_sizes[0] / D_TOT;   // 65536

  const size_t smem =
      (size_t)(K_CODES * CBN_STRIDE + D_M * CBT_STRIDE + WAVES * K_CODES * P_CSTRIDE)
          * sizeof(unsigned short)
      + (size_t)K_CODES * sizeof(float);   // ~231 KB, within 320 KB WGP LDS

  hipFuncSetAttribute(reinterpret_cast<const void*>(softpq_wmma_kernel),
                      hipFuncAttributeMaxDynamicSharedMemorySize, (int)smem);

  dim3 grid(n / (ROWS_PER_BLOCK * TILES), M_SUB);   // 128 x 8
  dim3 block(WAVES * 32);                           // 256 threads = 8 wave32
  softpq_wmma_kernel<<<grid, block, smem, stream>>>(x, cb, temp, out);
}
